// PaINNLayer_26216480374887
// MI455X (gfx1250) — matrix operand order repaired and checked by reading, not compiled, not run on hardware
//
#include <hip/hip_runtime.h>
#include <hip/hip_bf16.h>

// PaiNN-style equivariant GNN layer for MI455X (gfx1250, wave32, WMMA).
// Workspace layout (bytes), total 11,026,432:
//   [0,          524288)  conv_w2 packed bf16 B-fragments (256 tiles x 2 k x 32 lanes x 16)
//   [524288,     786432)  4 sq tensors packed bf16 B-fragments (w000,w110,w011,w101^T)
//   [786432,    2066432)  s_agg  (N x 32 f32)   -- segment_sum target (atomics)
//   [2066432,   5906432)  v_agg  (N x 32 x 3)
//   [5906432,   7186432)  s'     (N x 32)       -- after lin_conv
//   [7186432,  11026432)  v'     (N x 32 x 3)

#define MULC 32
#define NNODES 10000
#define NEDGES 40000

typedef __bf16 v16bf __attribute__((ext_vector_type(16)));
typedef float  v8f   __attribute__((ext_vector_type(8)));

__device__ __forceinline__ v8f wmma_bf16(v16bf a, v16bf b, v8f c) {
  // D = A(16x32 bf16) x B(32x16 bf16) + C(16x16 f32)
  return __builtin_amdgcn_wmma_f32_16x16x32_bf16(false, a, false, b, (short)0, c,
                                                 false, false);
}

__device__ __forceinline__ v8f splat8(float x) {
  v8f c;
#pragma unroll
  for (int r = 0; r < 8; ++r) c[r] = x;
  return c;
}

__device__ __forceinline__ float silu_f(float x) { return x / (1.0f + __expf(-x)); }

// Load 8 contiguous LDS floats (two b128) -- coefficient row for this lane.
__device__ __forceinline__ void load_cf8(const float* base, float cf[8]) {
  const float4* cp = (const float4*)base;
  const float4 c0 = cp[0];
  const float4 c1 = cp[1];
  cf[0] = c0.x; cf[1] = c0.y; cf[2] = c0.z; cf[3] = c0.w;
  cf[4] = c1.x; cf[5] = c1.y; cf[6] = c1.z; cf[7] = c1.w;
}

// One edge-kernel consumption pass over quadrant q of w = h@W2+b2.
// Tile t covers cols n=16t..16t+15: q=t>>6, i=(t>>1)&31, kh=t&1.
// coefT is [i][edge]-major (row stride 16): cf[r] = coefT[i*16 + hi*8 + r].
__device__ __forceinline__ void edge_pass(int q, const float* coefT,
                                          const v16bf a0, const v16bf a1,
                                          const v16bf* __restrict__ W2p,
                                          const float* __restrict__ cb2,
                                          int lane, int m, int hi, float acc[2][8]) {
  for (int i = 0; i < 32; ++i) {
    float cf[8];
    load_cf8(coefT + i * 16 + hi * 8, cf);
#pragma unroll
    for (int kh = 0; kh < 2; ++kh) {
      const int t = q * 64 + i * 2 + kh;
      const v16bf b0 = W2p[(t * 2 + 0) * 32 + lane];
      const v16bf b1 = W2p[(t * 2 + 1) * 32 + lane];
      v8f d = wmma_bf16(a0, b0, splat8(cb2[t * 16 + m]));
      d = wmma_bf16(a1, b1, d);
#pragma unroll
      for (int r = 0; r < 8; ++r) acc[kh][r] += cf[r] * d[r];
    }
  }
}

// One node-kernel consumption pass over packed sq matrix `mat` (32x1024).
__device__ __forceinline__ void node_pass(int mat, const float* coefT, float cmul,
                                          const v16bf a,
                                          const v16bf* __restrict__ SQp,
                                          int lane, int hi, float acc[2][8]) {
  const v8f zero = {0.f, 0.f, 0.f, 0.f, 0.f, 0.f, 0.f, 0.f};
  for (int jj = 0; jj < 32; ++jj) {
    float cf[8];
    load_cf8(coefT + jj * 16 + hi * 8, cf);
#pragma unroll
    for (int r = 0; r < 8; ++r) cf[r] *= cmul;
#pragma unroll
    for (int kh = 0; kh < 2; ++kh) {
      const int t = jj * 2 + kh;
      const v16bf b = SQp[(mat * 64 + t) * 32 + lane];
      const v8f d = wmma_bf16(a, b, zero);
#pragma unroll
      for (int r = 0; r < 8; ++r) acc[kh][r] += cf[r] * d[r];
    }
  }
}

// ---------------------------------------------------------------------------
__global__ void zero_kernel(float* __restrict__ p, int n) {
  int i = blockIdx.x * blockDim.x + threadIdx.x;
  if (i < n) p[i] = 0.0f;
}

// ---------------------------------------------------------------------------
// Pack conv_w2 (64 x 4096, K=64 rows) into bf16 WMMA B-fragment order:
// out[((t*2+kt)*32+lane)*16+j] = w2[K][N], K = kt*32+((j>>3)<<4)+(j&7)+(lane>>4)*8,
// N = t*16+(lane&15). A lane's 32-byte fragment is contiguous.
// ---------------------------------------------------------------------------
__global__ void pack_w2_kernel(const float* __restrict__ w2, __bf16* __restrict__ out) {
  int idx = blockIdx.x * blockDim.x + threadIdx.x;
  if (idx >= 262144) return;
  int j    = idx & 15;
  int lane = (idx >> 4) & 31;
  int kt   = (idx >> 9) & 1;
  int t    = idx >> 10;
  int K = kt * 32 + ((j >> 3) << 4) + (j & 7) + (lane >> 4) * 8;
  int N = t * 16 + (lane & 15);
  out[idx] = (__bf16)w2[K * 4096 + N];
}

// ---------------------------------------------------------------------------
// Pack the four (32,32,32) tensors as 32x1024 matrices in B-fragment order.
// mat 0: w000[i][jk], mat 1: w110[i][jk], mat 2: w011[i][jk],
// mat 3: w101 transposed -> B[j][(i,k)] = w101[i][j][k].
// ---------------------------------------------------------------------------
__global__ void pack_sq_kernel(const float* __restrict__ w000, const float* __restrict__ w110,
                               const float* __restrict__ w011, const float* __restrict__ w101,
                               __bf16* __restrict__ out) {
  int idx = blockIdx.x * blockDim.x + threadIdx.x;
  if (idx >= 131072) return;
  int mat = idx >> 15;
  int rem = idx & 32767;
  int j    = rem & 15;
  int lane = (rem >> 4) & 31;
  int t    = rem >> 9;                  // 0..63
  int Krow = ((j >> 3) << 4) + (j & 7) + (lane >> 4) * 8;  // 0..31
  int N    = t * 16 + (lane & 15);      // 0..1023
  float v;
  if (mat == 0)      v = w000[Krow * 1024 + N];
  else if (mat == 1) v = w110[Krow * 1024 + N];
  else if (mat == 2) v = w011[Krow * 1024 + N];
  else { int i = N >> 5, k = N & 31; v = w101[i * 1024 + Krow * 32 + k]; }
  out[idx] = (__bf16)v;
}

// ---------------------------------------------------------------------------
// Edge kernel: one wave = 16 edges. feats -> h (VALU) -> w = h@W2+b2 via
// bf16 WMMAs, consumed tile-by-tile. Passes are sequenced so only one
// 16-register accumulator set is live at a time (atomic scatter commutes).
// Coefficients stored [i][edge]-major so a lane's 8 fold coefficients are
// two ds_load_b128.
// ---------------------------------------------------------------------------
__global__ __launch_bounds__(32)
void edge_tp_kernel(const float* __restrict__ f, const float* __restrict__ esh,
                    const float* __restrict__ elen,
                    const float* __restrict__ cw1, const float* __restrict__ cb1,
                    const v16bf* __restrict__ W2p, const float* __restrict__ cb2,
                    const int* __restrict__ esrc, const int* __restrict__ edst,
                    float* __restrict__ s_agg, float* __restrict__ v_agg) {
  __shared__ float  feats[16][72];                 // [sdst(32) | ssrc(32) | elen(8)]
  __shared__ __bf16 h_l[16][64];
  __shared__ __align__(16) float syT[32 * 16];     // (ss*y0)[i][e]
  __shared__ __align__(16) float dotT[32 * 16];    // dot[i][e]
  __shared__ __align__(16) float ssT[32 * 16];     // ss[i][e]
  __shared__ __align__(16) float vyT[3][32 * 16];  // (vs_m*y0)[m][i][e]
  __shared__ float  y1_l[16][3];
  __shared__ int    dst_l[16];
  __shared__ float  w1_l[72 * 64];

  const int lane = threadIdx.x;
  const int e0   = blockIdx.x * 16;
  const int m    = lane & 15;
  const int hi   = lane >> 4;
  const float inv = 0.125f;  // (2*MUL)^-1/2

  for (int i = lane; i < 72 * 64; i += 32) w1_l[i] = cw1[i];

  for (int e = 0; e < 16; ++e) {
    const int ee  = e0 + e;
    const int src = esrc[ee];
    const int dst = edst[ee];
    if (lane == 0) dst_l[e] = dst;
    feats[e][lane]      = f[dst * 128 + lane];
    const float ss      = f[src * 128 + lane];
    feats[e][32 + lane] = ss;
    if (lane < 8) feats[e][64 + lane] = elen[ee * 8 + lane];
    const float y0  = esh[ee * 4 + 0];
    const float y1x = esh[ee * 4 + 1];
    const float y1y = esh[ee * 4 + 2];
    const float y1z = esh[ee * 4 + 3];
    if (lane < 3) y1_l[e][lane] = esh[ee * 4 + 1 + lane];
    const float vx = f[src * 128 + 32 + lane * 3 + 0];
    const float vy = f[src * 128 + 32 + lane * 3 + 1];
    const float vz = f[src * 128 + 32 + lane * 3 + 2];
    // lane == i index; transposed [i][e] layout
    dotT[lane * 16 + e]   = (vx * y1x + vy * y1y + vz * y1z) * 0.57735026918962576f;
    syT[lane * 16 + e]    = ss * y0;
    ssT[lane * 16 + e]    = ss;
    vyT[0][lane * 16 + e] = vx * y0;
    vyT[1][lane * 16 + e] = vy * y0;
    vyT[2][lane * 16 + e] = vz * y0;
  }
  __syncthreads();

  // h = silu(feats @ conv_w1 + b1), stored bf16
  for (int it = 0; it < 32; ++it) {
    const int linear = it * 32 + lane;
    const int e = linear >> 6;
    const int c = linear & 63;
    float acc = cb1[c];
    for (int r = 0; r < 72; ++r) acc += feats[e][r] * w1_l[r * 64 + c];
    h_l[e][c] = (__bf16)silu_f(acc);
  }
  __syncthreads();

  // A fragments (16 edges x 64 K) in registers, reused across all 256 N-tiles.
  v16bf a0, a1;
#pragma unroll
  for (int j = 0; j < 16; ++j) {
    const int kidx = ((j >> 3) << 4) + (j & 7) + hi * 8;
    a0[j] = h_l[m][kidx];
    a1[j] = h_l[m][kidx + 32];
  }

  // ---- s_tp: q=0 (coef ss*y0) + q=3 (coef dot), then scatter ----
  {
    float acc[2][8];
#pragma unroll
    for (int kh = 0; kh < 2; ++kh)
#pragma unroll
      for (int r = 0; r < 8; ++r) acc[kh][r] = 0.f;
    edge_pass(0, syT, a0, a1, W2p, cb2, lane, m, hi, acc);
    edge_pass(3, dotT, a0, a1, W2p, cb2, lane, m, hi, acc);
#pragma unroll
    for (int kh = 0; kh < 2; ++kh) {
      const int k = kh * 16 + m;
#pragma unroll
      for (int r = 0; r < 8; ++r) {
        const int dst = dst_l[hi * 8 + r];
        atomicAdd(&s_agg[dst * 32 + k], acc[kh][r] * inv);
      }
    }
  }

  // ---- v_tp part 1: q=1 (coef ss), scatter t01 * y1[m] ----
  {
    float acc[2][8];
#pragma unroll
    for (int kh = 0; kh < 2; ++kh)
#pragma unroll
      for (int r = 0; r < 8; ++r) acc[kh][r] = 0.f;
    edge_pass(1, ssT, a0, a1, W2p, cb2, lane, m, hi, acc);
#pragma unroll
    for (int kh = 0; kh < 2; ++kh) {
      const int k = kh * 16 + m;
#pragma unroll
      for (int r = 0; r < 8; ++r) {
        const int e   = hi * 8 + r;
        const int dst = dst_l[e];
        const float tv = acc[kh][r] * inv;
#pragma unroll
        for (int mm = 0; mm < 3; ++mm)
          atomicAdd(&v_agg[(dst * 32 + k) * 3 + mm], tv * y1_l[e][mm]);
      }
    }
  }

  // ---- v_tp part 2: q=2 (coef vs_m*y0), one pass per m component ----
#pragma unroll 1
  for (int mm = 0; mm < 3; ++mm) {
    float acc[2][8];
#pragma unroll
    for (int kh = 0; kh < 2; ++kh)
#pragma unroll
      for (int r = 0; r < 8; ++r) acc[kh][r] = 0.f;
    edge_pass(2, vyT[mm], a0, a1, W2p, cb2, lane, m, hi, acc);
#pragma unroll
    for (int kh = 0; kh < 2; ++kh) {
      const int k = kh * 16 + m;
#pragma unroll
      for (int r = 0; r < 8; ++r) {
        const int dst = dst_l[hi * 8 + r];
        atomicAdd(&v_agg[(dst * 32 + k) * 3 + mm], acc[kh][r] * inv);
      }
    }
  }
}

// ---------------------------------------------------------------------------
// Node lin_conv: s' = [s, s_agg] @ lin_conv_w0 * 0.125, v' likewise.
// ---------------------------------------------------------------------------
__global__ void node_pre_kernel(const float* __restrict__ f,
                                const float* __restrict__ s_agg, const float* __restrict__ v_agg,
                                const float* __restrict__ lw0, const float* __restrict__ lw1,
                                float* __restrict__ sp, float* __restrict__ vp) {
  const int idx = blockIdx.x * blockDim.x + threadIdx.x;
  if (idx >= NNODES * 32) return;
  const int n = idx >> 5;
  const int k = idx & 31;
  float acc = 0.f, a0 = 0.f, a1 = 0.f, a2 = 0.f;
  for (int i = 0; i < 64; ++i) {
    const float w0 = lw0[i * 32 + k];
    const float w1 = lw1[i * 32 + k];
    float sc, vx, vy, vz;
    if (i < 32) {
      sc = f[n * 128 + i];
      vx = f[n * 128 + 32 + i * 3 + 0];
      vy = f[n * 128 + 32 + i * 3 + 1];
      vz = f[n * 128 + 32 + i * 3 + 2];
    } else {
      const int ii = i - 32;
      sc = s_agg[n * 32 + ii];
      vx = v_agg[(n * 32 + ii) * 3 + 0];
      vy = v_agg[(n * 32 + ii) * 3 + 1];
      vz = v_agg[(n * 32 + ii) * 3 + 2];
    }
    acc += sc * w0; a0 += vx * w1; a1 += vy * w1; a2 += vz * w1;
  }
  sp[idx] = acc * 0.125f;
  vp[idx * 3 + 0] = a0 * 0.125f;
  vp[idx * 3 + 1] = a1 * 0.125f;
  vp[idx * 3 + 2] = a2 * 0.125f;
}

// ---------------------------------------------------------------------------
// Node squared-tensor + gate MLP + lin_upd + norms. One wave = 16 nodes.
// Coefficients consumed from transposed copies before in-place gating.
// ---------------------------------------------------------------------------
__global__ __launch_bounds__(32)
void node_sq_kernel(const float* __restrict__ sp, const float* __restrict__ vp,
                    const v16bf* __restrict__ SQp,
                    const float* __restrict__ uw1, const float* __restrict__ ub1,
                    const float* __restrict__ uw2, const float* __restrict__ ub2,
                    const float* __restrict__ luw0, const float* __restrict__ luw1,
                    const float* __restrict__ lnw, const float* __restrict__ lnb,
                    float* __restrict__ out) {
  __shared__ float sp_l[16][32];
  __shared__ float v_l[16][3][32];
  __shared__ __align__(16) float spT[32 * 16];    // s'[j][node]
  __shared__ __align__(16) float vT[3][32 * 16];  // v'[m][j][node]
  __shared__ float ssq_l[16][32];
  __shared__ float vsq_l[16][3][32];
  __shared__ float hid_l[16][64];

  const int lane = threadIdx.x;
  const int n0   = blockIdx.x * 16;
  const int m    = lane & 15;
  const int hi   = lane >> 4;
  const float sq_inv = 0.022097086912079612f;  // (2*32*32)^-1/2

  for (int e = 0; e < 16; ++e) {
    const int n = n0 + e;
    const float sv = sp[n * 32 + lane];
    sp_l[e][lane] = sv;
    spT[lane * 16 + e] = sv;
#pragma unroll
    for (int mm = 0; mm < 3; ++mm) {
      const float vv = vp[(n * 32 + lane) * 3 + mm];
      v_l[e][mm][lane] = vv;
      vT[mm][lane * 16 + e] = vv;
    }
  }
  __syncthreads();

  v16bf as, av0, av1, av2;
#pragma unroll
  for (int j = 0; j < 16; ++j) {
    const int kidx = ((j >> 3) << 4) + (j & 7) + hi * 8;
    as[j]  = (__bf16)sp_l[m][kidx];
    av0[j] = (__bf16)v_l[m][0][kidx];
    av1[j] = (__bf16)v_l[m][1][kidx];
    av2[j] = (__bf16)v_l[m][2][kidx];
  }

  // ---- s_sq: w000 (A=s, coef s_j) + w110 (A=v_m, coef v_jm * 3^-1/2) ----
  {
    float acc[2][8];
#pragma unroll
    for (int kh = 0; kh < 2; ++kh)
#pragma unroll
      for (int r = 0; r < 8; ++r) acc[kh][r] = 0.f;
    node_pass(0, spT, 1.0f, as, SQp, lane, hi, acc);
    node_pass(1, vT[0], 0.57735026918962576f, av0, SQp, lane, hi, acc);
    node_pass(1, vT[1], 0.57735026918962576f, av1, SQp, lane, hi, acc);
    node_pass(1, vT[2], 0.57735026918962576f, av2, SQp, lane, hi, acc);
#pragma unroll
    for (int kh = 0; kh < 2; ++kh)
#pragma unroll
      for (int r = 0; r < 8; ++r) ssq_l[hi * 8 + r][kh * 16 + m] = acc[kh][r] * sq_inv;
  }

  // ---- v_sq per m: w011 (A=s, coef v_jm) + w101^T (A=s, coef v_im) ----
#pragma unroll 1
  for (int mm = 0; mm < 3; ++mm) {
    float acc[2][8];
#pragma unroll
    for (int kh = 0; kh < 2; ++kh)
#pragma unroll
      for (int r = 0; r < 8; ++r) acc[kh][r] = 0.f;
    node_pass(2, vT[mm], 1.0f, as, SQp, lane, hi, acc);
    node_pass(3, vT[mm], 1.0f, as, SQp, lane, hi, acc);
#pragma unroll
    for (int kh = 0; kh < 2; ++kh)
#pragma unroll
      for (int r = 0; r < 8; ++r)
        vsq_l[hi * 8 + r][mm][kh * 16 + m] = acc[kh][r] * sq_inv;
  }
  __syncthreads();

  // gate MLP stage 1: hid = silu([s', s_sq] @ upd_w1 + b1)
  for (int it = 0; it < 32; ++it) {
    const int linear = it * 32 + lane;
    const int e = linear >> 6;
    const int c = linear & 63;
    float acc = ub1[c];
    for (int i = 0; i < 64; ++i) {
      const float sc = (i < 32) ? sp_l[e][i] : ssq_l[e][i - 32];
      acc += sc * uw1[i * 64 + c];
    }
    hid_l[e][c] = silu_f(acc);
  }
  __syncthreads();
  // gate MLP stage 2 + apply gate in place
  for (int it = 0; it < 32; ++it) {
    const int linear = it * 32 + lane;
    const int e = linear >> 6;
    const int c = linear & 63;
    float g = ub2[c];
    for (int i = 0; i < 64; ++i) g += hid_l[e][i] * uw2[i * 64 + c];
    if (c < 32) {
      sp_l[e][c] *= g;
#pragma unroll
      for (int mm = 0; mm < 3; ++mm) v_l[e][mm][c] *= g;
    } else {
      const int cc = c - 32;
      ssq_l[e][cc] *= g;
#pragma unroll
      for (int mm = 0; mm < 3; ++mm) vsq_l[e][mm][cc] *= g;
    }
  }
  __syncthreads();

  // lin_upd + layernorm(s) + vector-norm(v); lane owns output channel k
  const int k = lane;
  for (int e = 0; e < 16; ++e) {
    float s2 = 0.f, v20 = 0.f, v21 = 0.f, v22 = 0.f;
    for (int i = 0; i < 64; ++i) {
      const float w0 = luw0[i * 32 + k];
      const float w1 = luw1[i * 32 + k];
      float sc, vc0, vc1, vc2;
      if (i < 32) {
        sc = sp_l[e][i];
        vc0 = v_l[e][0][i]; vc1 = v_l[e][1][i]; vc2 = v_l[e][2][i];
      } else {
        const int ii = i - 32;
        sc = ssq_l[e][ii];
        vc0 = vsq_l[e][0][ii]; vc1 = vsq_l[e][1][ii]; vc2 = vsq_l[e][2][ii];
      }
      s2 += sc * w0; v20 += vc0 * w1; v21 += vc1 * w1; v22 += vc2 * w1;
    }
    s2 *= 0.125f; v20 *= 0.125f; v21 *= 0.125f; v22 *= 0.125f;

    float sum = s2, sumsq = s2 * s2;
    float vss = v20 * v20 + v21 * v21 + v22 * v22;
#pragma unroll
    for (int off = 16; off > 0; off >>= 1) {
      sum   += __shfl_xor(sum, off);
      sumsq += __shfl_xor(sumsq, off);
      vss   += __shfl_xor(vss, off);
    }
    const float mean = sum * (1.f / 32.f);
    const float var  = sumsq * (1.f / 32.f) - mean * mean;
    const float sout = (s2 - mean) * rsqrtf(var + 1e-5f) * lnw[k] + lnb[k];
    const float nv   = vss * (1.f / 96.f);
    const float vsc  = rsqrtf(nv + 1e-5f) * lnw[32 + k];
    const int n = n0 + e;
    out[n * 128 + k] = sout;
    out[n * 128 + 32 + k * 3 + 0] = v20 * vsc;
    out[n * 128 + 32 + k * 3 + 1] = v21 * vsc;
    out[n * 128 + 32 + k * 3 + 2] = v22 * vsc;
  }
}

// ---------------------------------------------------------------------------
extern "C" void kernel_launch(void* const* d_in, const int* in_sizes, int n_in,
                              void* d_out, int out_size, void* d_ws, size_t ws_size,
                              hipStream_t stream) {
  const float* f    = (const float*)d_in[0];
  const float* esh  = (const float*)d_in[1];
  const float* elen = (const float*)d_in[2];
  const float* cw1  = (const float*)d_in[3];
  const float* cb1  = (const float*)d_in[4];
  const float* cw2  = (const float*)d_in[5];
  const float* cb2  = (const float*)d_in[6];
  const float* lcw0 = (const float*)d_in[7];
  const float* lcw1 = (const float*)d_in[8];
  const float* w000 = (const float*)d_in[9];
  const float* w011 = (const float*)d_in[10];
  const float* w101 = (const float*)d_in[11];
  const float* w110 = (const float*)d_in[12];
  const float* uw1  = (const float*)d_in[13];
  const float* ub1  = (const float*)d_in[14];
  const float* uw2  = (const float*)d_in[15];
  const float* ub2  = (const float*)d_in[16];
  const float* luw0 = (const float*)d_in[17];
  const float* luw1 = (const float*)d_in[18];
  const float* lnw  = (const float*)d_in[19];
  const float* lnb  = (const float*)d_in[20];
  const int* esrc   = (const int*)d_in[21];
  const int* edst   = (const int*)d_in[22];
  float* out = (float*)d_out;

  char* ws = (char*)d_ws;
  const size_t OFF_W2P  = 0;
  const size_t OFF_SQP  = 524288;
  const size_t OFF_SAGG = 786432;
  const size_t OFF_VAGG = OFF_SAGG + (size_t)NNODES * 32 * 4;
  const size_t OFF_SP   = OFF_VAGG + (size_t)NNODES * 96 * 4;
  const size_t OFF_VP   = OFF_SP   + (size_t)NNODES * 32 * 4;

  __bf16* W2p  = (__bf16*)(ws + OFF_W2P);
  __bf16* SQp  = (__bf16*)(ws + OFF_SQP);
  float* s_agg = (float*)(ws + OFF_SAGG);
  float* v_agg = (float*)(ws + OFF_VAGG);
  float* sp    = (float*)(ws + OFF_SP);
  float* vp    = (float*)(ws + OFF_VP);

  // s_agg and v_agg are contiguous: zero both (N*128 floats).
  zero_kernel<<<(NNODES * 128 + 255) / 256, 256, 0, stream>>>(s_agg, NNODES * 128);
  pack_w2_kernel<<<262144 / 256, 256, 0, stream>>>(cw2, W2p);
  pack_sq_kernel<<<131072 / 256, 256, 0, stream>>>(w000, w110, w011, w101, SQp);

  edge_tp_kernel<<<NEDGES / 16, 32, 0, stream>>>(f, esh, elen, cw1, cb1,
                                                 (const v16bf*)W2p, cb2, esrc, edst,
                                                 s_agg, v_agg);
  node_pre_kernel<<<(NNODES * 32 + 255) / 256, 256, 0, stream>>>(f, s_agg, v_agg,
                                                                 lcw0, lcw1, sp, vp);
  node_sq_kernel<<<NNODES / 16, 32, 0, stream>>>(sp, vp, (const v16bf*)SQp,
                                                 uw1, ub1, uw2, ub2, luw0, luw1,
                                                 lnw, lnb, out);
  (void)in_sizes; (void)n_in; (void)out_size; (void)ws_size;
}
